// DPFABase_65996467470369
// MI455X (gfx1250) — matrix operand order repaired and checked by reading, not compiled, not run on hardware
//
#include <hip/hip_runtime.h>

typedef __attribute__((ext_vector_type(16))) _Float16 v16h;
typedef __attribute__((ext_vector_type(8)))  _Float16 v8h;
typedef __attribute__((ext_vector_type(4)))  _Float16 v4h;
typedef __attribute__((ext_vector_type(8)))  float    v8f;
typedef __attribute__((ext_vector_type(4)))  float    v4f;

#define BB 128
#define SS 512
#define HH 128

// ---------------------------------------------------------------------------
// Kernel 1: gather + L2-normalize embeddings -> f16, and gather mastery.
// One wave per (b,s) row; rows [0,BS) = history, [BS,2BS) = next.
// ---------------------------------------------------------------------------
__global__ __launch_bounds__(256) void dpfa_prep(
    const int* __restrict__ hist_items, const int* __restrict__ next_items,
    const int* __restrict__ corrects,   const float* __restrict__ emb,
    const float* __restrict__ resp,
    _Float16* __restrict__ hist16, _Float16* __restrict__ next16,
    float* __restrict__ mastery)
{
    const int lane = threadIdx.x & 31;
    const int row  = blockIdx.x * 8 + (threadIdx.x >> 5);
    const int BS   = BB * SS;
    const bool isHist = row < BS;
    const int bs   = isHist ? row : row - BS;
    const int item = isHist ? hist_items[bs] : next_items[bs];

    const v4f* src = (const v4f*)(emb + (size_t)item * HH);
    v4f v = src[lane];                       // 32 lanes x 4 floats = 128
    float ss = v[0]*v[0] + v[1]*v[1] + v[2]*v[2] + v[3]*v[3];
    #pragma unroll
    for (int k = 1; k < 32; k <<= 1) ss += __shfl_xor(ss, k, 32);
    const float inv = 1.0f / sqrtf(ss);

    _Float16* dst = (isHist ? hist16 : next16) + (size_t)bs * HH;
    v4h o;
    #pragma unroll
    for (int j = 0; j < 4; ++j) o[j] = (_Float16)(v[j] * inv);
    ((v4h*)dst)[lane] = o;

    if (isHist && lane == 0) {
        const int ic = (corrects[bs] == 2) ? 1 : 0;
        mastery[bs] = resp[item * 2 + ic];
    }
}

// ---------------------------------------------------------------------------
// Kernel 2: streaming causal attention via v_wmma_f32_16x16x32_f16.
// Each wave owns a 16-query tile. Double-buffered B tiles (next tile's
// 8x global_load_b128 issued before current tile's WMMAs), online softmax
// fused with the mastery reduction, diagonal tile peeled so the hot loop
// carries no causal-mask VALU.
// ---------------------------------------------------------------------------
__global__ __launch_bounds__(256) void dpfa_attn(
    const _Float16* __restrict__ hist16, const _Float16* __restrict__ next16,
    const float* __restrict__ mastery,   const int* __restrict__ hist_items,
    const int* __restrict__ next_items,  const float* __restrict__ beta,
    const float* __restrict__ tdk_p,     const float* __restrict__ tdb_p,
    float* __restrict__ out)
{
    const int lane = threadIdx.x & 31;
    const int wave = threadIdx.x >> 5;
    const int l16  = lane & 15;
    const int hf   = lane >> 4;                  // 16-lane half-group
    const int b    = blockIdx.y;
    const int q_base = blockIdx.x * 128 + wave * 16;
    const float tdk = *tdk_p, tdb = *tdb_p;
    const size_t rowbase = (size_t)b * SS;

    // --- A fragments: 4 chunks of K=32 covering H=128 (loaded once) ---
    // lane l16 = query row; elems 0..7 <- K = c*32 + hf*16 + j,
    //                       elems 8..15 <- K = c*32 + hf*16 + 8 + j.
    const _Float16* qptr = next16 + (rowbase + q_base + l16) * HH;
    v16h a[4];
    #pragma unroll
    for (int c = 0; c < 4; ++c) {
        v8h lo = *(const v8h*)(qptr + c * 32 + hf * 16);
        v8h hi = *(const v8h*)(qptr + c * 32 + hf * 16 + 8);
        #pragma unroll
        for (int j = 0; j < 8; ++j) { a[c][j] = lo[j]; a[c][8 + j] = hi[j]; }
    }

    // per-lane online softmax state: row r = q_base + hf*8 + r, col = l16
    float m[8], den[8], num[8], rowAdd[8];
    #pragma unroll
    for (int r = 0; r < 8; ++r) {
        m[r] = -3.0e38f; den[r] = 0.0f; num[r] = 0.0f;
        rowAdd[r] = tdk * (float)(q_base + hf * 8 + r);   // loop-invariant
    }

    auto loadB = [&](v16h (&bf)[4], int t) {
        // B frag: lane l16 = key column; elem j <- K = c*32 + hf*16 + j
        const _Float16* kptr = hist16 + (rowbase + (size_t)(t * 16 + l16)) * HH;
        #pragma unroll
        for (int c = 0; c < 4; ++c)
            bf[c] = *(const v16h*)(kptr + c * 32 + hf * 16);
    };

    auto tileScore = [&](const v16h (&bf)[4]) {
        v8f acc = {0.f, 0.f, 0.f, 0.f, 0.f, 0.f, 0.f, 0.f};
        #pragma unroll
        for (int c = 0; c < 4; ++c)
            acc = __builtin_amdgcn_wmma_f32_16x16x32_f16(
                false, a[c], false, bf[c], (short)0, acc, false, false);
        return acc;
    };

    auto update = [&](v8f acc, int t, bool diag) {
        const int key = t * 16 + l16;
        const float mast = mastery[rowbase + key];
        float dcol = tdb + tdk * (float)(1 - key);
        if (hist_items[rowbase + key] == 0) dcol += -1e9f;   // pad mask
        #pragma unroll
        for (int r = 0; r < 8; ++r) {
            float z = acc[r] + dcol + rowAdd[r];
            if (diag) {                                      // causal mask
                const int qr = q_base + hf * 8 + r;
                z += (key > qr) ? -1e9f : 0.0f;
            }
            const float mn = fmaxf(m[r], z);
            const float w  = __expf(fminf(m[r], z) - mn);    // single exp
            const bool  zb = z > m[r];
            const float sc = zb ? w : 1.0f;                  // rescale old
            const float p  = zb ? 1.0f : w;                  // new term
            den[r] = den[r] * sc + p;
            num[r] = num[r] * sc + p * mast;
            m[r]   = mn;
        }
    };

    // causal: tiles 0 .. ntiles-1; the last one is the diagonal tile.
    const int ntiles = q_base / 16 + 1;
    const int N = ntiles - 1;                    // # non-diagonal tiles
    v16h b0[4], b1[4];
    loadB(b0, 0);
    int t = 0;
    while (t + 2 <= N) {                         // pipelined pairs
        loadB(b1, t + 1);
        update(tileScore(b0), t, false);
        loadB(b0, t + 2);
        update(tileScore(b1), t + 1, false);
        t += 2;
    }
    if (t < N) {                                 // one leftover + diagonal
        loadB(b1, t + 1);                        // t+1 == N (diagonal)
        update(tileScore(b0), t, false);
        update(tileScore(b1), N, true);
    } else {                                     // diagonal already in b0
        update(tileScore(b0), N, true);
    }

    // merge partial softmax states across the 16 lanes of the half-group
    #pragma unroll
    for (int r = 0; r < 8; ++r) {
        #pragma unroll
        for (int k = 1; k < 16; k <<= 1) {
            const float om = __shfl_xor(m[r],   k, 32);
            const float od = __shfl_xor(den[r], k, 32);
            const float on = __shfl_xor(num[r], k, 32);
            const float mn = fmaxf(m[r], om);
            const float s0 = __expf(m[r] - mn);
            const float s1 = __expf(om   - mn);
            den[r] = den[r] * s0 + od * s1;
            num[r] = num[r] * s0 + on * s1;
            m[r]   = mn;
        }
    }

    if (l16 == 0) {
        #pragma unroll
        for (int r = 0; r < 8; ++r) {
            const int qr = q_base + hf * 8 + r;
            const float ability = num[r] / den[r];
            const float bt = beta[next_items[rowbase + qr]];
            out[rowbase + qr] = 1.0f / (1.0f + __expf(-(ability - bt)));
        }
    }
}

// ---------------------------------------------------------------------------
extern "C" void kernel_launch(void* const* d_in, const int* in_sizes, int n_in,
                              void* d_out, int out_size, void* d_ws, size_t ws_size,
                              hipStream_t stream) {
    (void)in_sizes; (void)n_in; (void)out_size; (void)ws_size;
    const int*   hist_items = (const int*)  d_in[0];
    const int*   next_items = (const int*)  d_in[1];
    const int*   corrects   = (const int*)  d_in[2];
    const float* emb        = (const float*)d_in[3];
    const float* beta       = (const float*)d_in[4];
    const float* resp       = (const float*)d_in[5];
    const float* tdk        = (const float*)d_in[6];
    const float* tdb        = (const float*)d_in[7];
    float*       out        = (float*)d_out;

    const size_t BS = (size_t)BB * SS;
    _Float16* hist16  = (_Float16*)d_ws;                // 16 MB
    _Float16* next16  = hist16 + BS * HH;               // 16 MB
    float*    mastery = (float*)(next16 + BS * HH);     // 256 KB

    // 2*B*S rows, 8 rows (waves) per 256-thread block
    dpfa_prep<<<dim3((unsigned)(2 * BS / 8)), 256, 0, stream>>>(
        hist_items, next_items, corrects, emb, resp, hist16, next16, mastery);

    // 8 waves/block * 16 queries = 128 queries per block
    dpfa_attn<<<dim3(SS / 128, BB), 256, 0, stream>>>(
        hist16, next16, mastery, hist_items, next_items, beta, tdk, tdb, out);
}
